// GeometricStructureEmbeddingV2_2173253452346
// MI455X (gfx1250) — matrix operand order repaired and checked by reading, not compiled, not run on hardware
//
#include <hip/hip_runtime.h>
#include <hip/hip_bf16.h>
#include <math.h>

typedef _Float16 v16h __attribute__((ext_vector_type(16)));
typedef _Float16 v8h  __attribute__((ext_vector_type(8)));
typedef float    v8f  __attribute__((ext_vector_type(8)));
typedef int      vi4  __attribute__((__vector_size__(16)));
typedef __attribute__((address_space(1))) vi4* gv4p;   // global v4i*
typedef __attribute__((address_space(3))) vi4* lv4p;   // LDS v4i*

#define NPTS   512
#define HID    128
#define WSTR   144   // weight row stride (halfs), 288B, 32B aligned
#define ASTR   144   // f16 activation stage row stride (halfs)
#define FSTR   132   // f32 stage row stride (floats)
#define LN10K_64 0.14391156831212787f  // ln(10000)/64
#define RAD2DEG  57.29577951308232f

#define WBYTES   (HID*WSTR*2)        // 36864 B per weight image
#define TBYTES   (181*HID*4)         // 92672 B angle table

// ------------------------------------------------------------ async copy ---
__device__ __forceinline__ void block_async_copy(const void* g, void* l, int bytes,
                                                 int tid, int nthr) {
#if __has_builtin(__builtin_amdgcn_global_load_async_to_lds_b128)
    const char* gp = (const char*)g;
    char*       lp = (char*)l;
    for (int off = tid * 16; off < bytes; off += nthr * 16) {
        __builtin_amdgcn_global_load_async_to_lds_b128(
            (gv4p)(unsigned long long)(uintptr_t)(gp + off),
            (lv4p)(unsigned int)(uintptr_t)(lp + off),
            0, 0);
    }
#else
    const float4* gp = (const float4*)g;
    float4*       lp = (float4*)l;
    int n16 = bytes >> 4;
    for (int i = tid; i < n16; i += nthr) lp[i] = gp[i];
#endif
}

__device__ __forceinline__ void wait_async_done() {
#if __has_builtin(__builtin_amdgcn_global_load_async_to_lds_b128)
#if __has_builtin(__builtin_amdgcn_s_wait_asynccnt)
    __builtin_amdgcn_s_wait_asynccnt(0);
#else
    asm volatile("s_wait_asynccnt 0x0" ::: "memory");
#endif
#endif
}

// ---------------------------------------------------------------- knn ------
__global__ void knn_kernel(const float* __restrict__ p, float* __restrict__ refv) {
    int n = blockIdx.x * blockDim.x + threadIdx.x;
    if (n >= NPTS) return;
    float px = p[n*3+0], py = p[n*3+1], pz = p[n*3+2];
    float bd2[4] = {1e30f, 1e30f, 1e30f, 1e30f};
    int   bi[4]  = {0, 0, 0, 0};
    for (int m = 0; m < NPTS; ++m) {
        float dx = p[m*3+0]-px, dy = p[m*3+1]-py, dz = p[m*3+2]-pz;
        float d2 = dx*dx + dy*dy + dz*dz;
        if (d2 < bd2[3]) {
            int j = 3;
            while (j > 0 && d2 < bd2[j-1]) { bd2[j] = bd2[j-1]; bi[j] = bi[j-1]; --j; }
            bd2[j] = d2; bi[j] = m;
        }
    }
    for (int k = 0; k < 3; ++k) {      // drop nearest (self)
        int m = bi[k+1];
        refv[(n*3+k)*3+0] = p[m*3+0]-px;
        refv[(n*3+k)*3+1] = p[m*3+1]-py;
        refv[(n*3+k)*3+2] = p[m*3+2]-pz;
    }
}

// ------------------------------------------------ angle table (181 x 128) --
__global__ void build_table(const float* __restrict__ Wa, const float* __restrict__ ba,
                            float* __restrict__ table) {
    __shared__ float emb[HID];
    int a = blockIdx.x;          // 0..180
    int c = threadIdx.x;         // 0..127
    float t  = (float)a * (1.0f / 15.0f);
    float om = t * __expf((float)(c >> 1) * -LN10K_64);
    emb[c] = (c & 1) ? __cosf(om) : __sinf(om);
    __syncthreads();
    float s = ba[c];
    #pragma unroll 4
    for (int k = 0; k < HID; ++k) s += emb[k] * Wa[k*HID + c];
    table[a*HID + c] = s;
}

// --------------------------- one-time transposed f16 weight image in d_ws --
__global__ void convert_weights(const float* __restrict__ Wd, const float* __restrict__ We,
                                _Float16* __restrict__ wdT, _Float16* __restrict__ weT) {
    int i = blockIdx.x * blockDim.x + threadIdx.x;
    if (i >= HID*HID) return;
    int k = i >> 7, c = i & 127;
    wdT[c*WSTR + k] = (_Float16)Wd[i];
    weT[c*WSTR + k] = (_Float16)We[i];
}

// -------------------------------------------------- fragment constructors --
// A (16x32 f16): lane L -> M = L&15 ; halves 0..7 = K base..base+7,
// halves 8..15 = K base+16..base+23, base = kc*32 + (L>=16)*8
__device__ __forceinline__ v16h load_A_frag(const _Float16* a16, int lane, int kc) {
    int M    = lane & 15;
    int base = kc*32 + (lane >> 4)*8;
    const _Float16* q = a16 + M*ASTR + base;
    v8h lo = *(const v8h*)(q);
    v8h hi = *(const v8h*)(q + 16);
    v16h r;
    #pragma unroll
    for (int i = 0; i < 8; ++i) { r[i] = lo[i]; r[8+i] = hi[i]; }
    return r;
}
// B (32x16 f16): lane L -> N = t*16 + (L&15) ; halves 0..7 = K base..base+7,
// halves 8..15 = K base+16..base+23, base = kc*32 + (L>=16)*8, from transposed weight wT[N][k]
__device__ __forceinline__ v16h load_B_frag(const _Float16* wT, int lane, int kc, int t) {
    int N    = t*16 + (lane & 15);
    int base = kc*32 + (lane >> 4)*8;
    const _Float16* q = wT + N*WSTR + base;
    v8h lo = *(const v8h*)(q);
    v8h hi = *(const v8h*)(q + 16);
    v16h r;
    #pragma unroll
    for (int i = 0; i < 8; ++i) { r[i] = lo[i]; r[8+i] = hi[i]; }
    return r;
}

// -------------------------------------------------------- fused main -------
__global__ __launch_bounds__(256)
void fused_kernel(const float* __restrict__ p,
                  const float* __restrict__ bd, const float* __restrict__ be,
                  const float* __restrict__ gamma, const float* __restrict__ beta,
                  const float* __restrict__ tableG, const float* __restrict__ refv,
                  const _Float16* __restrict__ wdG, const _Float16* __restrict__ weG,
                  float* __restrict__ out) {
    extern __shared__ char smem[];
    _Float16* wdT      = (_Float16*)smem;                       // 36864 B
    _Float16* weT      = wdT + HID*WSTR;                        // 36864 B
    float*    tableL   = (float*)(weT + HID*WSTR);              // 92672 B
    float*    stageAll = tableL + 181*HID;                      // 67584 B
    _Float16* a16All   = (_Float16*)(stageAll + 8*16*FSTR);     // 36864 B
    int*      aidxAll  = (int*)(a16All + 8*16*ASTR);            // 1536 B
    float*    drAll    = (float*)(aidxAll + 8*48);              // 512 B
    float*    muAll    = drAll + 8*16;                          // 512 B
    float*    rsAll    = muAll + 8*16;                          // 512 B
    float*    divt     = rsAll + 8*16;                          // 256 B
    float*    bdL      = divt + 64;                             // 512 B
    float*    beL      = bdL + HID;
    float*    gmL      = beL + HID;
    float*    btL      = gmL + HID;

    const int tid  = threadIdx.x;
    const int lane = tid & 31;
    const int wave = tid >> 5;
    const int n    = blockIdx.x >> 2;
    const int m0   = (blockIdx.x & 3) * 128 + wave * 16;

    // ---- async bulk copies: weights (f16 image) + angle table -------------
    block_async_copy(wdG,    wdT,    WBYTES, tid, 256);
    block_async_copy(weG,    weT,    WBYTES, tid, 256);
    block_async_copy(tableG, tableL, TBYTES, tid, 256);

    // ---- small LDS setup overlapped with async transfers ------------------
    if (tid < 64) divt[tid] = __expf((float)tid * -LN10K_64);
    if (tid < HID) {
        bdL[tid] = bd[tid];  beL[tid] = be[tid];
        gmL[tid] = gamma[tid]; btL[tid] = beta[tid];
    }

    float*    stagef = stageAll + wave*16*FSTR;
    _Float16* a16    = a16All   + wave*16*ASTR;
    int*      aidx   = aidxAll  + wave*48;
    float*    dr     = drAll    + wave*16;
    float*    mu     = muAll    + wave*16;
    float*    rs     = rsAll    + wave*16;

    const float pnx = p[n*3+0], pny = p[n*3+1], pnz = p[n*3+2];

    // per-row distance index + 3 angle indices (overlapped with async)
    if (lane < 16) {
        int m = m0 + lane;
        float ax = p[m*3+0]-pnx, ay = p[m*3+1]-pny, az = p[m*3+2]-pnz;
        dr[lane] = sqrtf(ax*ax + ay*ay + az*az) * 5.0f;   // /SIGMA_D
        #pragma unroll
        for (int k = 0; k < 3; ++k) {
            float rx = refv[(n*3+k)*3+0], ry = refv[(n*3+k)*3+1], rz = refv[(n*3+k)*3+2];
            float cx = ry*az - rz*ay;
            float cy = rz*ax - rx*az;
            float cz = rx*ay - ry*ax;
            float sv = sqrtf(cx*cx + cy*cy + cz*cz);
            float cv = rx*ax + ry*ay + rz*az;
            int a = (int)rintf(atan2f(sv, cv) * RAD2DEG);
            a = min(180, max(0, a));
            aidx[lane*3 + k] = a;
        }
    }

    wait_async_done();
    __syncthreads();

    // a_emb max (from LDS table) into f32 stage ; distance embedding -> f16
    for (int i = lane; i < 16*HID; i += 32) {
        int r = i >> 7, c = i & 127;
        int a0 = aidx[r*3+0], a1 = aidx[r*3+1], a2 = aidx[r*3+2];
        float amax = fmaxf(tableL[a0*HID+c], fmaxf(tableL[a1*HID+c], tableL[a2*HID+c]));
        stagef[r*FSTR + c] = amax;
        float om = dr[r] * divt[c >> 1];
        float ev = (c & 1) ? __cosf(om) : __sinf(om);
        a16[r*ASTR + c] = (_Float16)ev;
    }
    __syncthreads();

    // ---- GEMM1: emb(16x128 f16) @ Wd -> acc (f32) -------------------------
    v16h afrag[4];
    #pragma unroll
    for (int kc = 0; kc < 4; ++kc) afrag[kc] = load_A_frag(a16, lane, kc);

    v8f acc[8];
    #pragma unroll
    for (int t = 0; t < 8; ++t) {
        v8f c8 = {};
        #pragma unroll
        for (int kc = 0; kc < 4; ++kc) {
            v16h b = load_B_frag(wdT, lane, kc, t);
            c8 = __builtin_amdgcn_wmma_f32_16x16x32_f16(false, afrag[kc], false, b,
                                                        (short)0, c8, false, false);
        }
        acc[t] = c8;
    }

    // ---- + bd + a_emb, leaky relu, restage as f16 -------------------------
    {
        const int colb = lane & 15;
        const int mhi  = (lane >> 4) * 8;
        #pragma unroll
        for (int t = 0; t < 8; ++t) {
            int c = t*16 + colb;
            float bdc = bdL[c];
            #pragma unroll
            for (int i = 0; i < 8; ++i) {
                int M = mhi + i;
                float e = acc[t][i] + bdc + stagef[M*FSTR + c];
                e = (e >= 0.0f) ? e : 0.2f * e;
                a16[M*ASTR + c] = (_Float16)e;
            }
        }
    }
    __syncthreads();

    // ---- GEMM2: e(16x128 f16) @ We -> acc ---------------------------------
    #pragma unroll
    for (int kc = 0; kc < 4; ++kc) afrag[kc] = load_A_frag(a16, lane, kc);

    #pragma unroll
    for (int t = 0; t < 8; ++t) {
        v8f c8 = {};
        #pragma unroll
        for (int kc = 0; kc < 4; ++kc) {
            v16h b = load_B_frag(weT, lane, kc, t);
            c8 = __builtin_amdgcn_wmma_f32_16x16x32_f16(false, afrag[kc], false, b,
                                                        (short)0, c8, false, false);
        }
        acc[t] = c8;
    }

    // ---- + be, stage f32 for layernorm ------------------------------------
    {
        const int colb = lane & 15;
        const int mhi  = (lane >> 4) * 8;
        #pragma unroll
        for (int t = 0; t < 8; ++t) {
            int c = t*16 + colb;
            float bec = beL[c];
            #pragma unroll
            for (int i = 0; i < 8; ++i) {
                int M = mhi + i;
                stagef[M*FSTR + c] = acc[t][i] + bec;
            }
        }
    }
    __syncthreads();

    if (lane < 16) {
        float s = 0.0f, s2 = 0.0f;
        #pragma unroll 4
        for (int c = 0; c < HID; ++c) {
            float v = stagef[lane*FSTR + c];
            s += v; s2 += v*v;
        }
        float m_  = s * (1.0f/HID);
        float var = s2 * (1.0f/HID) - m_*m_;
        mu[lane] = m_;
        rs[lane] = rsqrtf(var + 1e-5f);
    }
    __syncthreads();

    // ---- layernorm apply + vectorized b128 stores -------------------------
    const long base = ((long)(n*NPTS + m0)) * HID;
    for (int idx = lane; idx < 16*32; idx += 32) {   // 32 float4 per row
        int r  = idx >> 5;
        int c4 = (idx & 31) * 4;
        float m_ = mu[r], rr = rs[r];
        float4 v;
        v.x = (stagef[r*FSTR + c4+0] - m_) * rr * gmL[c4+0] + btL[c4+0];
        v.y = (stagef[r*FSTR + c4+1] - m_) * rr * gmL[c4+1] + btL[c4+1];
        v.z = (stagef[r*FSTR + c4+2] - m_) * rr * gmL[c4+2] + btL[c4+2];
        v.w = (stagef[r*FSTR + c4+3] - m_) * rr * gmL[c4+3] + btL[c4+3];
        *(float4*)(out + base + (long)r*HID + c4) = v;
    }
}

// ---------------------------------------------------------------------------
extern "C" void kernel_launch(void* const* d_in, const int* in_sizes, int n_in,
                              void* d_out, int out_size, void* d_ws, size_t ws_size,
                              hipStream_t stream) {
    const float* p     = (const float*)d_in[0];
    const float* Wd    = (const float*)d_in[1];
    const float* bd    = (const float*)d_in[2];
    const float* Wa    = (const float*)d_in[3];
    const float* ba    = (const float*)d_in[4];
    const float* We    = (const float*)d_in[5];
    const float* be    = (const float*)d_in[6];
    const float* gamma = (const float*)d_in[7];
    const float* beta  = (const float*)d_in[8];
    float* out = (float*)d_out;

    // workspace: [table 181*128 f32][refv 512*9 f32][wdT f16 img][weT f16 img]
    float*     table = (float*)d_ws;
    float*     refv  = table + 181*HID;
    char*      wbase = (char*)(refv + NPTS*9);
    uintptr_t  walign = (((uintptr_t)wbase + 31) & ~(uintptr_t)31) - (uintptr_t)wbase;
    _Float16*  wdT   = (_Float16*)(wbase + walign);
    _Float16*  weT   = (_Float16*)((char*)wdT + WBYTES);

    knn_kernel     <<<(NPTS + 63)/64, 64, 0, stream>>>(p, refv);
    build_table    <<<181, HID, 0, stream>>>(Wa, ba, table);
    convert_weights<<<(HID*HID + 255)/256, 256, 0, stream>>>(Wd, We, wdT, weT);

    size_t smem = (size_t)2*WBYTES                 // wdT + weT f16 images
                + (size_t)TBYTES                   // angle table
                + (size_t)(8*16*FSTR)*sizeof(float)
                + (size_t)(8*16*ASTR)*sizeof(_Float16)
                + (size_t)(8*48)*sizeof(int)
                + (size_t)(3*8*16)*sizeof(float)   // dr, mu, rs
                + (size_t)64*sizeof(float)         // div_term
                + (size_t)4*HID*sizeof(float);     // bd, be, gamma, beta
    fused_kernel<<<NPTS*4, 256, smem, stream>>>(p, bd, be, gamma, beta,
                                                table, refv, wdT, weT, out);
}
